// VerticalSelfAttention_39256001085944
// MI455X (gfx1250) — matrix operand order repaired and checked by reading, not compiled, not run on hardware
//
#include <hip/hip_runtime.h>
#include <hip/hip_bf16.h>

typedef __attribute__((ext_vector_type(16))) __bf16 v16bf;
typedef __attribute__((ext_vector_type(8)))  float  v8f;

#define Bc 32
#define Cc 64
#define Rc 63
#define Ec 256
#define Sc 64
#define Hc 4
#define Dc 64

#define SSTR 264   // seq/k LDS row stride in bf16 elems (64 rows x 256 + 8 pad)
#define VSTR 72    // vT LDS row stride (256 rows x 64 + 8 pad)

#define WKV_ELEMS (2 * Ec * Ec)   // K + V weight matrices, bf16, in d_ws

union FragBf {
    v16bf v;
    unsigned short u[16];
    uint4 q[2];
};

static __device__ __forceinline__ unsigned short f2bf(float f) {
    unsigned int u = __float_as_uint(f);
    u += 0x7fffu + ((u >> 16) & 1u);   // round-to-nearest-even
    return (unsigned short)(u >> 16);
}
static __device__ __forceinline__ float bf2f(unsigned short h) {
    return __uint_as_float(((unsigned int)h) << 16);
}

// ---- Prep kernel: convert K/V weight rows (w_in rows 256..767) to bf16 once.
__global__ __launch_bounds__(256) void prep_wkv(const float* __restrict__ w_in,
                                                unsigned short* __restrict__ wkv) {
    const int stride = gridDim.x * blockDim.x;
    int i4 = blockIdx.x * blockDim.x + threadIdx.x;
    const float4* src = (const float4*)(w_in + Ec * Ec);   // skip Q rows
    for (; i4 < WKV_ELEMS / 4; i4 += stride) {
        float4 v = src[i4];
        unsigned short* dst = wkv + i4 * 4;
        dst[0] = f2bf(v.x); dst[1] = f2bf(v.y); dst[2] = f2bf(v.z); dst[3] = f2bf(v.w);
    }
}

__global__ __launch_bounds__(256) void vsa_kernel(
    const float* __restrict__ x, const int* __restrict__ real_cols,
    const float* __restrict__ cls_token,
    const float* __restrict__ w_in, const float* __restrict__ b_in,
    const float* __restrict__ w_out, const float* __restrict__ b_out,
    const float* __restrict__ ln_g, const float* __restrict__ ln_b,
    const float* __restrict__ w1, const float* __restrict__ b1,
    const float* __restrict__ w2, const float* __restrict__ b2,
    const unsigned short* __restrict__ wkv,
    float* __restrict__ out)
{
    __shared__ unsigned short seq[Sc * SSTR];    // bf16 seq (cls + x rows)
    __shared__ unsigned short klds[Sc * SSTR];   // bf16 K  [s][e]
    __shared__ unsigned short vT[Ec * VSTR];     // bf16 V^T [e][s]
    __shared__ float q0[Ec];
    __shared__ float scores[Hc * Sc];
    __shared__ float probs[Hc * Sc];
    __shared__ float ctx0[Ec];
    __shared__ float aout[Ec];
    __shared__ float ln0[Ec];
    __shared__ float hid[Ec];

    const int tid  = threadIdx.x;
    const int lane = tid & 31;
    const int wv   = tid >> 5;
    const int bc   = blockIdx.x;           // 0 .. B*C-1
    const int bIdx = bc >> 6;              // / C
    const int cIdx = bc & 63;              // % C

    // ---- Stage 0: build seq (row0 = cls token, rows 1..63 = x) as bf16 in LDS
    for (int it = tid; it < (Sc * Ec) / 4; it += 256) {
        int elem = it * 4;
        int row = elem >> 8;
        int col = elem & 255;
        float4 vv;
        if (row == 0)
            vv = *(const float4*)(cls_token + col);
        else
            vv = *(const float4*)(x + ((size_t)bc * Rc + (row - 1)) * Ec + col);
        unsigned short* sp = seq + row * SSTR + col;
        sp[0] = f2bf(vv.x); sp[1] = f2bf(vv.y); sp[2] = f2bf(vv.z); sp[3] = f2bf(vv.w);
    }

    // ---- Stage 2 (independent): q for CLS row only, full f32 GEMV
    {
        float s = b_in[tid];
        const float4* wr = (const float4*)(w_in + (size_t)tid * Ec);
        const float4* cl = (const float4*)cls_token;
        #pragma unroll 4
        for (int k = 0; k < Ec / 4; ++k) {
            float4 a = cl[k], w = wr[k];
            s += a.x * w.x + a.y * w.y + a.z * w.z + a.w * w.w;
        }
        q0[tid] = s;
    }
    __syncthreads();

    // ---- Stage 1: K and V projections via WMMA bf16 (D = seq @ W^T + b)
    {
        const int mat = wv >> 2;                         // 0 = K, 1 = V
        const unsigned short* wbase = wkv + (size_t)mat * Ec * Ec;
        const float* bbase = b_in + (mat + 1) * Ec;
        const int ng   = wv & 3;
        const int half = lane >> 4;
        const int l16  = lane & 15;

        // Branchless output addressing (K: row-major, V: transposed)
        unsigned short* const obase = (mat == 0) ? klds : vT;
        const int rstride = (mat == 0) ? SSTR : 1;     // multiplies sequence pos
        const int cstride = (mat == 0) ? 1 : VSTR;     // multiplies feature col

        for (int nt0 = 0; nt0 < 4; ++nt0) {
            const int nt   = ng * 4 + nt0;
            const int ncol = nt * 16 + l16;
            v8f acc[4];
            #pragma unroll
            for (int m = 0; m < 4; ++m)
                #pragma unroll
                for (int r = 0; r < 8; ++r) acc[m][r] = 0.0f;

            for (int ks = 0; ks < 8; ++ks) {
                const int k0 = ks * 32;
                // B fragment: bf16 weight row `ncol`, K-contiguous (preconverted)
                FragBf bf;
                const unsigned short* wp = wbase + (size_t)ncol * Ec + k0 + 16 * half;
                bf.q[0] = *(const uint4*)(wp);
                bf.q[1] = *(const uint4*)(wp + 8);

                #pragma unroll
                for (int m = 0; m < 4; ++m) {
                    // A fragment: seq rows, two 16B K-chunks per lane
                    FragBf af;
                    const unsigned short* ap =
                        seq + (m * 16 + l16) * SSTR + k0 + 16 * half;
                    af.q[0] = *(const uint4*)(ap);
                    af.q[1] = *(const uint4*)(ap + 8);
                    acc[m] = __builtin_amdgcn_wmma_f32_16x16x32_bf16(
                        false, af.v, false, bf.v, (short)0, acc[m], false, false);
                }
            }

            const float bias = bbase[ncol];
            unsigned short* const ocol = obase + ncol * cstride;
            #pragma unroll
            for (int m = 0; m < 4; ++m) {
                #pragma unroll
                for (int r = 0; r < 8; ++r) {
                    const int row = m * 16 + r + 8 * half;   // sequence position
                    ocol[row * rstride] = f2bf(acc[m][r] + bias);
                }
            }
        }
    }
    __syncthreads();

    // ---- Stage 3: scores[h][j] = (q0_h . k_j) / sqrt(D)   (CLS query only)
    {
        const int h = tid >> 6, j = tid & 63;
        const unsigned short* kr = klds + j * SSTR + h * Dc;
        const float* qh = q0 + h * Dc;
        float s = 0.0f;
        #pragma unroll 8
        for (int d = 0; d < Dc; ++d) s += qh[d] * bf2f(kr[d]);
        scores[tid] = s * 0.125f;
    }
    __syncthreads();

    // ---- softmax over the 64 keys of this thread's head
    {
        const int h = tid >> 6;
        const float* sr = scores + h * Sc;
        float mx = -3.0e38f;
        for (int j = 0; j < Sc; ++j) mx = fmaxf(mx, sr[j]);
        float sum = 0.0f;
        for (int j = 0; j < Sc; ++j) sum += expf(sr[j] - mx);
        probs[tid] = expf(scores[tid] - mx) / sum;
    }
    __syncthreads();

    // ---- Stage 4: ctx0[e] = sum_j probs[h][j] * V[j][e]  (V^T rows contiguous)
    {
        const int h = tid >> 6;
        const unsigned short* vr = vT + tid * VSTR;
        const float* pr = probs + h * Sc;
        float s = 0.0f;
        #pragma unroll 8
        for (int j = 0; j < Sc; ++j) s += pr[j] * bf2f(vr[j]);
        ctx0[tid] = s;
    }
    __syncthreads();

    // ---- Stage 5: attn_out0 = ctx0 @ w_out^T + b_out
    {
        float s = b_out[tid];
        const float4* wr = (const float4*)(w_out + (size_t)tid * Ec);
        #pragma unroll 4
        for (int k = 0; k < Ec / 4; ++k) {
            float4 w = wr[k];
            s += w.x * ctx0[4 * k] + w.y * ctx0[4 * k + 1] +
                 w.z * ctx0[4 * k + 2] + w.w * ctx0[4 * k + 3];
        }
        aout[tid] = s;
    }
    __syncthreads();

    // ---- LayerNorm over E (broadcast LDS reads)
    {
        float sm = 0.0f, s2 = 0.0f;
        for (int k = 0; k < Ec; ++k) { float a = aout[k]; sm += a; s2 += a * a; }
        const float mu  = sm * (1.0f / Ec);
        const float var = s2 * (1.0f / Ec) - mu * mu;
        const float inv = rsqrtf(var + 1e-5f);
        ln0[tid] = (aout[tid] - mu) * inv * ln_g[tid] + ln_b[tid];
    }
    __syncthreads();

    // ---- FFN layer 1: relu(ln0 @ w1^T + b1)
    {
        float s = b1[tid];
        const float4* wr = (const float4*)(w1 + (size_t)tid * Ec);
        #pragma unroll 4
        for (int k = 0; k < Ec / 4; ++k) {
            float4 w = wr[k];
            s += w.x * ln0[4 * k] + w.y * ln0[4 * k + 1] +
                 w.z * ln0[4 * k + 2] + w.w * ln0[4 * k + 3];
        }
        hid[tid] = fmaxf(s, 0.0f);
    }
    __syncthreads();

    // ---- FFN layer 2 + residual + mask + store
    {
        float s = b2[tid];
        const float4* wr = (const float4*)(w2 + (size_t)tid * Ec);
        #pragma unroll 4
        for (int k = 0; k < Ec / 4; ++k) {
            float4 w = wr[k];
            s += w.x * hid[4 * k] + w.y * hid[4 * k + 1] +
                 w.z * hid[4 * k + 2] + w.w * hid[4 * k + 3];
        }
        const float outv = ln0[tid] + s;
        const bool live = cIdx < real_cols[bIdx];
        out[(size_t)bc * Ec + tid] = live ? outv : 0.0f;
    }
}

extern "C" void kernel_launch(void* const* d_in, const int* in_sizes, int n_in,
                              void* d_out, int out_size, void* d_ws, size_t ws_size,
                              hipStream_t stream) {
    const float* x         = (const float*)d_in[0];
    const int*   real_cols = (const int*)d_in[1];
    const float* cls_token = (const float*)d_in[2];
    const float* w_in      = (const float*)d_in[3];
    const float* b_in      = (const float*)d_in[4];
    const float* w_out     = (const float*)d_in[5];
    const float* b_out     = (const float*)d_in[6];
    const float* ln_g      = (const float*)d_in[7];
    const float* ln_b      = (const float*)d_in[8];
    const float* w1        = (const float*)d_in[9];
    const float* b1        = (const float*)d_in[10];
    const float* w2        = (const float*)d_in[11];
    const float* b2        = (const float*)d_in[12];
    float* outp            = (float*)d_out;
    unsigned short* wkv    = (unsigned short*)d_ws;   // 256 KB bf16 K/V weights

    prep_wkv<<<dim3(128), dim3(256), 0, stream>>>(w_in, wkv);
    vsa_kernel<<<dim3(Bc * Cc), dim3(256), 0, stream>>>(
        x, real_cols, cls_token, w_in, b_in, w_out, b_out,
        ln_g, ln_b, w1, b1, w2, b2, wkv, outp);
}
